// LSTMModel_66700842107330
// MI455X (gfx1250) — matrix operand order repaired and checked by reading, not compiled, not run on hardware
//
#include <hip/hip_runtime.h>

typedef __attribute__((ext_vector_type(16))) _Float16 v16h;
typedef __attribute__((ext_vector_type(8)))  _Float16 v8h;
typedef __attribute__((ext_vector_type(8)))  float    v8f;
typedef __attribute__((ext_vector_type(4)))  float    v4f;

#define BATCH 4096
#define TSTEP 512
#define HID   50
#define G4    200   // 4*H gate rows
#define NP    208   // padded gate columns = 13*16
#define KP    64    // padded K: 0..49 = h, 50 = x (per step), 51 = 1.0 (bias row), 52..63 = 0
#define NT    13    // number of 16-wide N tiles

// CDNA5 has a hardware v_tanh_f32 transcendental — use it if the builtin exists.
#if __has_builtin(__builtin_amdgcn_tanhf)
__device__ __forceinline__ float tanh_hw(float x) { return __builtin_amdgcn_tanhf(x); }
#else
__device__ __forceinline__ float tanh_hw(float x) { return 1.0f - 2.0f / (__expf(2.0f * x) + 1.0f); }
#endif
// sigmoid(x) = 0.5 + 0.5*tanh(x/2): 1 trans op instead of exp+rcp (2 trans ops)
__device__ __forceinline__ float sigf(float x) { return fmaf(0.5f, tanh_hw(0.5f * x), 0.5f); }

struct BTile { v16h k0, k1; };   // B operands for the two K=32 steps of one N tile

// One wave (32 lanes) owns a 16-row batch tile; fully wave-internal (no barriers).
// Weights live in LDS as f16, gate-interleaved (n = 4*h + g) and K-augmented so a
// single WMMA chain computes  h@W_hh^T + x*W_ih^T + (b_ih+b_hh)  from C=0.
__global__ __launch_bounds__(32)
void lstm_wmma_kernel(const float* __restrict__ x,
                      const float* __restrict__ W_ih,
                      const float* __restrict__ W_hh,
                      const float* __restrict__ b_ih,
                      const float* __restrict__ b_hh,
                      const float* __restrict__ W_fc,
                      const float* __restrict__ b_fc,
                      float* __restrict__ out)
{
    __shared__ __align__(16) _Float16 wlds[NP * KP];   // [n][k] augmented W, f16
    __shared__ __align__(16) float    gatesl[16 * NP]; // [batch row][gate col] f32
    __shared__ __align__(16) _Float16 h16[16 * KP];    // [batch row][k] augmented activations
    __shared__ __align__(16) float    plds[32];

    const int lane = threadIdx.x;   // 0..31
    const int l2   = lane & 15;
    const int hi   = lane >> 4;
    const int b0   = blockIdx.x * 16;

    // ---- Stage augmented weights: k<50 -> W_hh, k==50 -> W_ih, k==51 -> b_ih+b_hh
    for (int e = lane; e < NP * KP; e += 32) {
        int n = e / KP, k = e % KP;
        float v = 0.0f;
        if (n < G4) {
            int g = n & 3, hh = n >> 2;
            int r = g * HID + hh;                 // original torch gate-major row
            if      (k <  HID)     v = W_hh[r * HID + k];
            else if (k == HID)     v = W_ih[r];   // I == 1
            else if (k == HID + 1) v = b_ih[r] + b_hh[r];
        }
        wlds[e] = (_Float16)v;
    }
    // ---- Activations: h = 0, x slot = 0, bias slot = 1.0, rest 0
    for (int e = lane; e < 16 * KP; e += 32)
        h16[e] = ((e % KP) == HID + 1) ? (_Float16)1.0f : (_Float16)0.0f;

    float c[25];
    #pragma unroll
    for (int q = 0; q < 25; ++q) c[q] = 0.0f;

    // per-lane B base address inside wlds: lane = column N, contiguous K run per half-wave
    const int wb0 = l2 * KP + hi * 16;

    auto loadB = [&](int ti) -> BTile {
        const int wbase = ti * (16 * KP) + wb0;
        union { v16h v; v8h h[2]; } u0, u1;
        u0.h[0] = *(const v8h*)&wlds[wbase];
        u0.h[1] = *(const v8h*)&wlds[wbase + 8];
        u1.h[0] = *(const v8h*)&wlds[wbase + 32];
        u1.h[1] = *(const v8h*)&wlds[wbase + 40];
        BTile r; r.k0 = u0.v; r.k1 = u1.v; return r;
    };

    const float* xrow = x + (b0 + l2) * TSTEP;   // lanes 16..31 duplicate 0..15 (harmless)
    float xv = xrow[0];                          // prefetched x for t=0

    for (int t = 0; t < TSTEP; ++t) {
        // publish current x into the augmented K slot, prefetch next step's x
        h16[l2 * KP + HID] = (_Float16)xv;
        int tn = (t + 1 < TSTEP) ? (t + 1) : (TSTEP - 1);
        xv = xrow[tn];

        // A operands (16x32 f16 layout): row = l2, contiguous K run of 16 per half-wave
        v16h a0, a1;
        {
            union { v16h v; v8h h[2]; } ua;
            ua.h[0] = *(const v8h*)&h16[l2 * KP +      hi * 16];
            ua.h[1] = *(const v8h*)&h16[l2 * KP + 8  + hi * 16];
            a0 = ua.v;
            ua.h[0] = *(const v8h*)&h16[l2 * KP + 32 + hi * 16];
            ua.h[1] = *(const v8h*)&h16[l2 * KP + 40 + hi * 16];
            a1 = ua.v;
        }

        // Software-pipelined N-tile loop: load tile ti+1's B while tile ti's WMMAs run
        BTile cur = loadB(0);
        v8f zero = {};
        #pragma unroll
        for (int ti = 0; ti < NT; ++ti) {
            BTile nxt = cur;
            if (ti + 1 < NT) nxt = loadB(ti + 1);

            v8f acc = __builtin_amdgcn_wmma_f32_16x16x32_f16(false, a0, false, cur.k0,
                                                             (short)0, zero, false, false);
            acc = __builtin_amdgcn_wmma_f32_16x16x32_f16(false, a1, false, cur.k1,
                                                         (short)0, acc, false, false);
            // D layout scatter: VGPR j -> batch row j + hi*8, column = ti*16 + l2
            #pragma unroll
            for (int j = 0; j < 8; ++j)
                gatesl[(j + hi * 8) * NP + ti * 16 + l2] = acc[j];

            cur = nxt;
        }

        // Elementwise cell update: lane owns (b = l2, h = 2q + hi); gates contiguous (i,f,g,o)
        #pragma unroll
        for (int q = 0; q < 25; ++q) {
            int hidx = 2 * q + hi;
            v4f g4 = *(const v4f*)&gatesl[l2 * NP + 4 * hidx];
            float ig = sigf(g4[0]);
            float fg = sigf(g4[1]);
            float gg = tanh_hw(g4[2]);
            float og = sigf(g4[3]);
            float cn = fg * c[q] + ig * gg;
            c[q] = cn;
            float hn = og * tanh_hw(cn);
            h16[l2 * KP + hidx] = (_Float16)hn;
        }
    }

    // Final projection: out[b] = sum_h h[b,h] * W_fc[h] + b_fc   (O == 1)
    float part = 0.0f;
    #pragma unroll
    for (int q = 0; q < 25; ++q) {
        int hidx = 2 * q + hi;
        part += (float)h16[l2 * KP + hidx] * W_fc[hidx];
    }
    plds[lane] = part;
    if (hi == 0) out[b0 + l2] = plds[l2] + plds[l2 + 16] + b_fc[0];
}

extern "C" void kernel_launch(void* const* d_in, const int* in_sizes, int n_in,
                              void* d_out, int out_size, void* d_ws, size_t ws_size,
                              hipStream_t stream) {
    (void)in_sizes; (void)n_in; (void)out_size; (void)d_ws; (void)ws_size;
    const float* x    = (const float*)d_in[0];
    const float* W_ih = (const float*)d_in[1];
    const float* W_hh = (const float*)d_in[2];
    const float* b_ih = (const float*)d_in[3];
    const float* b_hh = (const float*)d_in[4];
    const float* W_fc = (const float*)d_in[5];
    const float* b_fc = (const float*)d_in[6];
    float* out = (float*)d_out;
    // 256 independent single-wave workgroups (one 16-row batch tile each)
    lstm_wmma_kernel<<<BATCH / 16, 32, 0, stream>>>(x, W_ih, W_hh, b_ih, b_hh, W_fc, b_fc, out);
}
